// Net_83064667505278
// MI455X (gfx1250) — matrix operand order repaired and checked by reading, not compiled, not run on hardware
//
#include <hip/hip_runtime.h>
#include <hip/hip_bf16.h>

typedef _Float16 v16h __attribute__((ext_vector_type(16)));
typedef float    v8f  __attribute__((ext_vector_type(8)));

// ---------------- degree / norm ----------------

__global__ void k_init_deg(float* __restrict__ deg, int N) {
    int i = blockIdx.x * blockDim.x + threadIdx.x;
    if (i < N) deg[i] = 1.0f;                       // self-loop weight
}

__global__ void k_deg_scatter(const int* __restrict__ col,
                              const float* __restrict__ w,
                              float* __restrict__ deg, int E) {
    int e = blockIdx.x * blockDim.x + threadIdx.x;
    if (e >= E) return;
    __hip_atomic_fetch_add(&deg[col[e]], w[e],
                           __ATOMIC_RELAXED, __HIP_MEMORY_SCOPE_AGENT);
}

__global__ void k_dinv(const float* __restrict__ deg,
                       float* __restrict__ dinv, int N) {
    int i = blockIdx.x * blockDim.x + threadIdx.x;
    if (i < N) dinv[i] = rsqrtf(deg[i]);            // deg >= 1 always
}

// ---------------- GEMM1: h1 = x @ W1   (K=9 padded to 32, Nout=32) ----------------
// One wave per full 16-row tile, no bounds checks (tail handled separately).
// A (16-bit 16x32, ISA 7.12.2): lanes 0-15: M=lane, halves0-7=K0-7, halves8-15=K16-23;
//                               lanes16-31: M=lane-16, halves0-7=K8-15, halves8-15=K24-31.
// B (32x16): lanes 0-15: K0-7,K16-23 (2 halves/VGPR), lanes16-31: K8-15,K24-31 (same order as A); N=lane%16.

__global__ void k_gemm1_wmma(const float* __restrict__ x,
                             const float* __restrict__ W1,
                             float* __restrict__ h1, int n_tiles) {
    int wave = (blockIdx.x * blockDim.x + threadIdx.x) >> 5;
    int lane = threadIdx.x & 31;
    if (wave >= n_tiles) return;
    int row0 = wave * 16;
    int m    = lane & 15;
    int hi   = lane >> 4;

    // Every lane loads its row's 9 floats unconditionally (cached, branch-free).
    const float* xr = x + (size_t)(row0 + m) * 9;
    float xv[9];
    #pragma unroll
    for (int k = 0; k < 9; ++k) xv[k] = xr[k];

    v16h a;
    // hi==0: halves0-7 = K0..7 ; hi==1: halves0-7 = K8..15 (only K8 nonzero).
    a[0] = (_Float16)(hi ? xv[8] : xv[0]);
    #pragma unroll
    for (int i = 1; i < 8; ++i) a[i] = hi ? (_Float16)0.f : (_Float16)xv[i];
    #pragma unroll
    for (int i = 8; i < 16; ++i) a[i] = (_Float16)0.f;   // K16..31 all zero

    #pragma unroll
    for (int nt = 0; nt < 2; ++nt) {
        int n = nt * 16 + m;
        v16h b;
        #pragma unroll
        for (int i = 0; i < 16; ++i) {
            // table order: i<8 -> K=8*hi+i (valid K<9); i>=8 -> K>=16 (all zero).
            int kk = (hi ? 8 : 0) + i;
            float wv = (i < 8 && kk < 9) ? W1[kk * 32 + n] : 0.f;
            b[i] = (_Float16)wv;
        }
        v8f c = {};
        c = __builtin_amdgcn_wmma_f32_16x16x32_f16(false, a, false, b,
                                                   (short)0, c, false, false);
        float* outp = h1 + (size_t)(row0 + (hi ? 8 : 0)) * 32 + n;
        #pragma unroll
        for (int j = 0; j < 8; ++j) outp[(size_t)j * 32] = c[j];
    }
}

// scalar tail for rows not covered by full tiles (not used when N%16==0)
__global__ void k_gemm1_tail(const float* __restrict__ x,
                             const float* __restrict__ W1,
                             float* __restrict__ h1, int row_start, int N) {
    int t = blockIdx.x * blockDim.x + threadIdx.x;
    int row = row_start + (t >> 5), n = t & 31;
    if (row >= N) return;
    float acc = 0.f;
    #pragma unroll
    for (int k = 0; k < 9; ++k) acc += x[(size_t)row * 9 + k] * W1[k * 32 + n];
    h1[(size_t)row * 32 + n] = acc;
}

// ---------------- layer-1 aggregation ----------------

__global__ void k_agg1_init(const float* __restrict__ dinv,
                            const float* __restrict__ h1,
                            const float* __restrict__ b1,
                            float* __restrict__ agg1, int N) {
    int t = blockIdx.x * blockDim.x + threadIdx.x;     // N*32 threads
    if (t >= N * 32) return;
    int node = t >> 5, f = t & 31;
    float di = dinv[node];
    agg1[t] = di * di * h1[t] + b1[f];                 // self-loop + bias
}

__global__ void k_agg1_scatter(const int* __restrict__ row,
                               const int* __restrict__ col,
                               const float* __restrict__ w,
                               const float* __restrict__ dinv,
                               const float* __restrict__ h1,
                               float* __restrict__ agg1, int E) {
    // one wave per edge: lane = feature; edge state is wave-uniform -> scalar path
    int e = (blockIdx.x * blockDim.x + threadIdx.x) >> 5;
    int f = threadIdx.x & 31;
    if (e >= E) return;
    e = __builtin_amdgcn_readfirstlane(e);             // force SGPR / s_load path
    int r = row[e], c = col[e];
    float nrm = dinv[r] * w[e] * dinv[c];
    float v = nrm * h1[(size_t)r * 32 + f];            // coalesced 128B gather (L2)
    __hip_atomic_fetch_add(&agg1[(size_t)c * 32 + f], v,
                           __ATOMIC_RELAXED, __HIP_MEMORY_SCOPE_AGENT);
}

// ---------------- GEMM2: h2 = relu(agg1) @ W2  (K=32 native, Nout=7, stride 16) ----

__global__ void k_gemm2_wmma(const float* __restrict__ agg1,
                             const float* __restrict__ W2,
                             float* __restrict__ h2, int n_tiles) {
    int wave = (blockIdx.x * blockDim.x + threadIdx.x) >> 5;
    int lane = threadIdx.x & 31;
    if (wave >= n_tiles) return;
    int row0 = wave * 16;
    int m    = lane & 15;
    int hi   = lane >> 4;
    int kb1  = hi ? 8 : 0;      // halves 0..7
    int kb2  = hi ? 24 : 16;    // halves 8..15

    const float* ar = agg1 + (size_t)(row0 + m) * 32;
    v16h a;
    #pragma unroll
    for (int i = 0; i < 8; ++i) {
        float v = ar[kb1 + i]; v = v > 0.f ? v : 0.f;  // fused ReLU
        a[i] = (_Float16)v;
    }
    #pragma unroll
    for (int i = 0; i < 8; ++i) {
        float v = ar[kb2 + i]; v = v > 0.f ? v : 0.f;
        a[8 + i] = (_Float16)v;
    }

    int kbB = hi ? 16 : 0;
    int nc  = m < 7 ? m : 6;                           // clamped in-range address
    v16h b;
    #pragma unroll
    for (int i = 0; i < 16; ++i) {
        float wv = W2[((i < 8 ? kb1 : kb2 - 8) + i) * 7 + nc];             // unconditional load
        b[i] = (m < 7) ? (_Float16)wv : (_Float16)0.f; // cndmask, no branch
    }
    v8f c = {};
    c = __builtin_amdgcn_wmma_f32_16x16x32_f16(false, a, false, b,
                                               (short)0, c, false, false);
    // padded stride-16 store: unconditional for all 32 lanes
    float* outp = h2 + (size_t)(row0 + (hi ? 8 : 0)) * 16 + m;
    #pragma unroll
    for (int j = 0; j < 8; ++j) outp[(size_t)j * 16] = c[j];
}

__global__ void k_gemm2_tail(const float* __restrict__ agg1,
                             const float* __restrict__ W2,
                             float* __restrict__ h2, int row_start, int N) {
    int t = blockIdx.x * blockDim.x + threadIdx.x;
    int row = row_start + (t >> 3), n = t & 7;
    if (row >= N || n >= 7) return;
    float acc = 0.f;
    #pragma unroll
    for (int k = 0; k < 32; ++k) {
        float v = agg1[(size_t)row * 32 + k]; v = v > 0.f ? v : 0.f;
        acc += v * W2[k * 7 + n];
    }
    h2[(size_t)row * 16 + n] = acc;
}

// ---------------- layer-2 aggregation (h2 stride 16, agg2 stride 7) ----------------

__global__ void k_agg2_init(const float* __restrict__ dinv,
                            const float* __restrict__ h2,
                            const float* __restrict__ b2,
                            float* __restrict__ agg2, int N) {
    int t = blockIdx.x * blockDim.x + threadIdx.x;     // N*8 threads
    int node = t >> 3, f = t & 7;
    if (node >= N || f >= 7) return;
    float di = dinv[node];
    agg2[(size_t)node * 7 + f] = di * di * h2[(size_t)node * 16 + f] + b2[f];
}

__global__ void k_agg2_scatter(const int* __restrict__ row,
                               const int* __restrict__ col,
                               const float* __restrict__ w,
                               const float* __restrict__ dinv,
                               const float* __restrict__ h2,
                               float* __restrict__ agg2, int E) {
    int t = blockIdx.x * blockDim.x + threadIdx.x;     // E*8 threads
    int e = t >> 3, f = t & 7;
    if (e >= E || f >= 7) return;
    int r = row[e], c = col[e];
    float nrm = dinv[r] * w[e] * dinv[c];
    float v = nrm * h2[(size_t)r * 16 + f];
    __hip_atomic_fetch_add(&agg2[(size_t)c * 7 + f], v,
                           __ATOMIC_RELAXED, __HIP_MEMORY_SCOPE_AGENT);
}

// ---------------- log_softmax over 7 logits ----------------

__global__ void k_logsoftmax(const float* __restrict__ agg2,
                             float* __restrict__ out, int N) {
    int i = blockIdx.x * blockDim.x + threadIdx.x;
    if (i >= N) return;
    float v[7];
    float mx = -3.402823e38f;
    #pragma unroll
    for (int f = 0; f < 7; ++f) { v[f] = agg2[(size_t)i * 7 + f]; mx = fmaxf(mx, v[f]); }
    float s = 0.0f;
    #pragma unroll
    for (int f = 0; f < 7; ++f) s += __expf(v[f] - mx);
    float l = __logf(s);
    #pragma unroll
    for (int f = 0; f < 7; ++f) out[(size_t)i * 7 + f] = v[f] - mx - l;
}

// ---------------- launcher ----------------

extern "C" void kernel_launch(void* const* d_in, const int* in_sizes, int n_in,
                              void* d_out, int out_size, void* d_ws, size_t ws_size,
                              hipStream_t stream) {
    const float* x   = (const float*)d_in[0];
    const int*   ei  = (const int*)  d_in[1];
    const float* ew  = (const float*)d_in[2];
    const float* W1  = (const float*)d_in[3];
    const float* b1  = (const float*)d_in[4];
    const float* W2  = (const float*)d_in[5];
    const float* b2  = (const float*)d_in[6];
    float* out = (float*)d_out;

    const int N = in_sizes[0] / 9;
    const int E = in_sizes[2];
    const int* row = ei;           // edge_index[0]
    const int* col = ei + E;       // edge_index[1]

    // scratch carve (floats): deg N | dinv N | h1 32N | agg1 32N | h2 16N | agg2 7N
    float* ws   = (float*)d_ws;
    float* deg  = ws;
    float* dinv = deg  + N;
    float* h1   = dinv + N;
    float* agg1 = h1   + (size_t)N * 32;
    float* h2   = agg1 + (size_t)N * 32;
    float* agg2 = h2   + (size_t)N * 16;

    const int B = 256;
    // symmetric normalization  D^-1/2 (A+I) D^-1/2
    k_init_deg   <<<(N + B - 1) / B, B, 0, stream>>>(deg, N);
    k_deg_scatter<<<(E + B - 1) / B, B, 0, stream>>>(col, ew, deg, E);
    k_dinv       <<<(N + B - 1) / B, B, 0, stream>>>(deg, dinv, N);

    // h1 = x @ W1 via v_wmma_f32_16x16x32_f16 (full tiles, guard-free)
    int tiles = N / 16, rem = N % 16;
    int gw = (tiles * 32 + 127) / 128;
    k_gemm1_wmma<<<gw, 128, 0, stream>>>(x, W1, h1, tiles);
    if (rem) k_gemm1_tail<<<(rem * 32 + B - 1) / B, B, 0, stream>>>(x, W1, h1, tiles * 16, N);

    // agg1 = self-loop + b1, then edge scatter-add (wave per edge, scalarized edge state)
    k_agg1_init   <<<((size_t)N * 32 + B - 1) / B, B, 0, stream>>>(dinv, h1, b1, agg1, N);
    k_agg1_scatter<<<((size_t)E * 32 + B - 1) / B, B, 0, stream>>>(row, col, ew, dinv, h1, agg1, E);

    // h2 = relu(agg1) @ W2 via WMMA (K=32 native), padded stride-16 output
    k_gemm2_wmma<<<gw, 128, 0, stream>>>(agg1, W2, h2, tiles);
    if (rem) k_gemm2_tail<<<(rem * 8 + B - 1) / B, B, 0, stream>>>(agg1, W2, h2, tiles * 16, N);

    // agg2 = self-loop + b2, then edge scatter-add
    k_agg2_init   <<<((size_t)N * 8 + B - 1) / B, B, 0, stream>>>(dinv, h2, b2, agg2, N);
    k_agg2_scatter<<<((size_t)E * 8 + B - 1) / B, B, 0, stream>>>(row, col, ew, dinv, h2, agg2, E);

    // log_softmax
    k_logsoftmax<<<(N + B - 1) / B, B, 0, stream>>>(agg2, out, N);
}
